// MiniPSMNet_13580686590339
// MI455X (gfx1250) — compile-verified, matrix operand order checked by reading, not run on hardware
//
#include <hip/hip_runtime.h>
#include <hip/hip_bf16.h>

typedef __attribute__((ext_vector_type(16))) __bf16 v16bf;
typedef __attribute__((ext_vector_type(8)))  float  v8f;
typedef unsigned int u32x4 __attribute__((ext_vector_type(4)));
typedef int          i32x8 __attribute__((ext_vector_type(8)));
typedef int          i32x4 __attribute__((ext_vector_type(4)));

// ---------------------------------------------------------------------------
// Conv parameter block
// ---------------------------------------------------------------------------
struct ConvP {
    int N, Cin, Din, Hin, Win;
    int Cout, Do_, Ho, Wo;
    int G, coutg, cing;
    int Kreal, Ksteps, Ntiles;
    int pd, ph, pw;
    long long M;
};

// ---------------------------------------------------------------------------
// Pack f32 weights (OIDHW, grouped) into bf16 B-fragment panels:
// layout [g][ntile][kstep][lane 0..31][elem 0..15] so each lane's v16bf is a
// contiguous 32-byte chunk. K mapping follows the CDNA5 16-bit operand layout
// (lanes 0-15: K 0-7/16-23; lanes 16-31: K 8-15/24-31).
// ---------------------------------------------------------------------------
__global__ void pack_w_kernel(const float* __restrict__ w, __bf16* __restrict__ panel,
                              int G, int coutg, int cing, int kvol,
                              int kd, int kh, int kw,
                              int Kreal, int Ksteps, int Ntiles, int total)
{
    int idx = blockIdx.x * 256 + threadIdx.x;
    if (idx >= total) return;
    int e    = idx & 15;
    int rest = idx >> 4;
    int lane = rest & 31;  rest >>= 5;
    int s    = rest % Ksteps; rest /= Ksteps;
    int nt   = rest % Ntiles;
    int g    = rest / Ntiles;

    int n     = nt * 16 + (lane & 15);
    int kbase = (lane >> 4) * 8;
    int j     = e >> 1;
    int koff  = (j < 4 ? kbase + 2 * j : 16 + kbase + 2 * (j - 4)) + (e & 1);
    int k     = s * 32 + koff;

    float v = 0.f;
    if (k < Kreal && n < coutg) {
        int ci = k / kvol, rr = k % kvol;
        int tw = rr % kw, th = (rr / kw) % kh, td = rr / (kw * kh);
        int co = g * coutg + n;
        v = w[(((long long)(co * cing + ci) * kd + td) * kh + th) * kw + tw];
    }
    panel[idx] = (__bf16)v;
}

// ---------------------------------------------------------------------------
// Implicit-GEMM grouped conv, v_wmma_f32_16x16x32_bf16, B panel staged into
// LDS by the Tensor Data Mover (one DMA per workgroup, s_wait_tensorcnt).
// Block = 128 threads = 4 wave32; each wave owns one 16x16 C tile.
// Geometry is compile-time (KD,KH,KW,STRIDE,DIL) so tap decomposition is
// strength-reduced; DIL=1 implements the lhs-dilated-by-2 deconv (pad_lo=1).
// ---------------------------------------------------------------------------
template<int KD, int KH, int KW, int STRIDE, int DIL>
__global__ __launch_bounds__(128)
void conv_wmma_kernel(const float* __restrict__ X, const __bf16* __restrict__ Bp,
                      const float* __restrict__ bias, float* __restrict__ Y, ConvP p)
{
    constexpr int KVOL = KD * KH * KW;
    // Max panel: 27 k-steps * 512 bf16 = 27648 bytes
    __shared__ __attribute__((aligned(64))) __bf16 lpan[27 * 512];

    const int lane = threadIdx.x & 31;
    const int wave = threadIdx.x >> 5;
    const int g    = blockIdx.y;
    const int nt   = blockIdx.z;

    const __bf16* bpan = Bp + (long long)(g * p.Ntiles + nt) * p.Ksteps * 512;

    // ---- TDM: async-stage this block's B panel into LDS ------------------
    if (threadIdx.x < 32) {          // one TDM issue per block (EXEC ignored)
        unsigned lds_addr = (unsigned)(unsigned long long)&lpan[0]; // low 32b = LDS offset
        unsigned long long ga = (unsigned long long)bpan;
        unsigned units = (unsigned)p.Ksteps * 128u;   // panel size in 8-byte units
        // D# group0: count=1 | lds_addr | global_addr[56:0] | type=2
        u32x4 g0 = { 1u,
                     lds_addr,
                     (unsigned)(ga & 0xffffffffu),
                     (unsigned)((ga >> 32) & 0x01ffffffu) | (2u << 30) };
        // D# group1: data_size=3 (8B); tensor_dim0 = tile_dim0 = units;
        // tensor_dim1 = 1; 1-row tile; dim0_stride = units.
        i32x8 g1 = { (int)(3u << 16),
                     (int)((units & 0xffffu) << 16),
                     (int)((units >> 16) | (1u << 16)),
                     (int)((units & 0xffffu) << 16),
                     0,
                     (int)units,
                     0, 0 };
        i32x4 gz  = { 0, 0, 0, 0 };
        i32x8 gz8 = { 0, 0, 0, 0, 0, 0, 0, 0 };
        __builtin_amdgcn_tensor_load_to_lds(g0, g1, gz, gz, gz8, 0);
        __builtin_amdgcn_s_wait_tensorcnt(0);
    }
    __syncthreads();

    const long long m0 = ((long long)blockIdx.x * 4 + wave) * 16;
    const int HW  = p.Ho * p.Wo;
    const int DHW = p.Do_ * HW;

    // A-row this lane gathers (lanes 0-15 & 16-31 both cover rows 0-15)
    const int  rowA = lane & 15;
    const long long mA = m0 + rowA;
    const bool mvalid = (mA < p.M);
    int nb = 0, od = 0, oh = 0, ow = 0;
    if (mvalid) {
        nb = (int)(mA / DHW);
        int r = (int)(mA % DHW);
        od = r / HW; r %= HW;
        oh = r / p.Wo; ow = r % p.Wo;
    }
    const int kbase = (lane >> 4) * 8;

    const float* xg = X + (long long)(nb * p.Cin + g * p.cing) *
                          ((long long)p.Din * p.Hin * p.Win);
    const long long HinWin = (long long)p.Hin * p.Win;

    v8f acc = {};
    for (int s = 0; s < p.Ksteps; ++s) {
        v16bf bf = *(const v16bf*)&lpan[(s * 32 + lane) * 16];

        v16bf af;
#pragma unroll
        for (int e = 0; e < 16; ++e) {
            int j    = e >> 1;
            int koff = (j < 4 ? kbase + 2 * j : 16 + kbase + 2 * (j - 4)) + (e & 1);
            int k    = s * 32 + koff;
            float v = 0.f;
            if (mvalid && k < p.Kreal) {
                int ci = k / KVOL, rr = k % KVOL;               // KVOL constexpr
                int tw = rr % KW, th = (rr / KW) % KH, td = rr / (KW * KH);
                int id, ih, iw; bool ok;
                if (DIL == 0) {
                    id = od * STRIDE - p.pd + td;
                    ih = oh * STRIDE - p.ph + th;
                    iw = ow * STRIDE - p.pw + tw;
                    ok = (unsigned)id < (unsigned)p.Din &&
                         (unsigned)ih < (unsigned)p.Hin &&
                         (unsigned)iw < (unsigned)p.Win;
                } else {  // fractionally strided: input dilated by 2, pad_lo = 1
                    int dd = od - p.pd + td, hh = oh - p.ph + th, wi = ow - p.pw + tw;
                    ok = dd >= 0 && hh >= 0 && wi >= 0 &&
                         !(dd & 1) && !(hh & 1) && !(wi & 1);
                    id = dd >> 1; ih = hh >> 1; iw = wi >> 1;
                    ok = ok && id < p.Din && ih < p.Hin && iw < p.Win;
                }
                if (ok)
                    v = xg[((long long)ci * p.Din + id) * HinWin +
                           (long long)ih * p.Win + iw];
            }
            af[e] = (__bf16)v;
        }
        acc = __builtin_amdgcn_wmma_f32_16x16x32_bf16(
            false, af, false, bf, (short)0, acc, false, false);
    }

    // C layout: lane holds column (lane&15), rows r + 8*(lane>>4)
    const int ncol = nt * 16 + (lane & 15);
    if (ncol < p.coutg) {
        const int co = g * p.coutg + ncol;
        const float bv = bias ? bias[co] : 0.f;
#pragma unroll
        for (int rI = 0; rI < 8; ++rI) {
            long long mm = m0 + rI + ((lane >> 4) * 8);
            if (mm < p.M) {
                int nb2 = (int)(mm / DHW);
                int r   = (int)(mm % DHW);
                int od2 = r / HW; r %= HW;
                int oh2 = r / p.Wo, ow2 = r % p.Wo;
                Y[(((long long)nb2 * p.Cout + co) * p.Do_ + od2) * HW +
                  (long long)oh2 * p.Wo + ow2] = acc[rI] + bv;
            }
        }
    }
}

// ---------------------------------------------------------------------------
// BatchNorm: per-channel sum/sumsq reduce, then fused normalize+affine+ReLU.
// ---------------------------------------------------------------------------
__global__ void bn_stats_kernel(const float* __restrict__ X, float* __restrict__ stats,
                                int C, int N, long long DHW)
{
    __shared__ float ssum[256], ssq[256];
    const int c = blockIdx.x;
    const long long perC = (long long)N * DHW;
    float s = 0.f, q = 0.f;
    for (long long i = threadIdx.x; i < perC; i += 256) {
        int nb = (int)(i / DHW);
        long long off = i % DHW;
        float v = X[((long long)nb * C + c) * DHW + off];
        s += v; q += v * v;
    }
    ssum[threadIdx.x] = s; ssq[threadIdx.x] = q;
    __syncthreads();
    for (int k = 128; k > 0; k >>= 1) {
        if (threadIdx.x < k) {
            ssum[threadIdx.x] += ssum[threadIdx.x + k];
            ssq[threadIdx.x]  += ssq[threadIdx.x + k];
        }
        __syncthreads();
    }
    if (threadIdx.x == 0) { stats[2 * c] = ssum[0]; stats[2 * c + 1] = ssq[0]; }
}

__global__ void bn_apply_relu_kernel(float* __restrict__ X, const float* __restrict__ stats,
                                     const float* __restrict__ gamma,
                                     const float* __restrict__ beta,
                                     int C, long long DHW, long long total)
{
    long long i = (long long)blockIdx.x * 256 + threadIdx.x;
    if (i >= total) return;
    int c = (int)((i / DHW) % C);
    float perC = (float)(total / C);
    float mean = stats[2 * c] / perC;
    float var  = stats[2 * c + 1] / perC - mean * mean;
    float sc   = rsqrtf(var + 1e-5f) * gamma[c];
    float v = (X[i] - mean) * sc + beta[c];
    X[i] = v > 0.f ? v : 0.f;
}

// ---------------------------------------------------------------------------
// Cost volume: x is (4,16,96,96); batches 0..1 left, 2..3 right.
// cost is (2,32,32,96,96), channel = 2*c + {0:left, 1:right shifted}.
// ---------------------------------------------------------------------------
__global__ void cost_volume_kernel(const float* __restrict__ x, float* __restrict__ cost)
{
    long long i = (long long)blockIdx.x * 256 + threadIdx.x;
    if (i >= 18874368LL) return;
    int wI = (int)(i % 96);
    long long t = i / 96;
    int h = (int)(t % 96); t /= 96;
    int j = (int)(t % 32); t /= 32;
    int ch = (int)(t % 32);
    int nb = (int)(t / 32);
    int c = ch >> 1, side = ch & 1;
    float v = 0.f;
    if (wI >= j) {
        if (side == 0) v = x[(((long long)nb * 16 + c) * 96 + h) * 96 + wI];
        else           v = x[((((long long)nb + 2) * 16 + c) * 96 + h) * 96 + (wI - j)];
    }
    cost[i] = v;
}

__global__ void add_kernel(float* __restrict__ y, const float* __restrict__ s, long long n)
{
    long long i = (long long)blockIdx.x * 256 + threadIdx.x;
    if (i < n) y[i] += s[i];
}

__global__ void to2d_kernel(const float* __restrict__ u3, const float* __restrict__ w,
                            const float* __restrict__ b, float* __restrict__ out)
{
    int i = blockIdx.x * 256 + threadIdx.x;
    if (i >= 589824) return;
    int hw = i % 9216;
    int d  = (i / 9216) % 32;
    int nb = i / (9216 * 32);
    float s = b[0];
#pragma unroll
    for (int c = 0; c < 32; ++c)
        s += u3[(((long long)nb * 32 + c) * 32 + d) * 9216 + hw] * w[c];
    out[i] = s;
}

__global__ void predict_kernel(const float* __restrict__ o, const float* __restrict__ w,
                               const float* __restrict__ b, float* __restrict__ pred)
{
    int i = blockIdx.x * 256 + threadIdx.x;
    if (i >= 18432) return;
    int hw = i % 9216, nb = i / 9216;
    float s = b[0];
#pragma unroll
    for (int d = 0; d < 32; ++d)
        s += o[((long long)nb * 32 + d) * 9216 + hw] * w[d];
    pred[i] = s > 0.f ? s : 0.f;
}

// ---------------------------------------------------------------------------
// Host orchestration
// ---------------------------------------------------------------------------
static inline int idiv(long long a, long long b) { return (int)((a + b - 1) / b); }

extern "C" void kernel_launch(void* const* d_in, const int* in_sizes, int n_in,
                              void* d_out, int out_size, void* d_ws, size_t ws_size,
                              hipStream_t stream)
{
    (void)in_sizes; (void)n_in; (void)out_size; (void)ws_size;
    const float* conv_out = (const float*)d_in[0];
    const float* r_w1 = (const float*)d_in[1];
    const float* r_b1 = (const float*)d_in[2];
    const float* r_w2 = (const float*)d_in[3];
    const float* r_b2 = (const float*)d_in[4];
    const float* r_g  = (const float*)d_in[5];
    const float* r_bt = (const float*)d_in[6];
    const float* S[5][6];
    for (int s = 0; s < 5; ++s)
        for (int k = 0; k < 6; ++k)
            S[s][k] = (const float*)d_in[7 + s * 6 + k];
    const float* d1w = (const float*)d_in[37];
    const float* d2w = (const float*)d_in[38];
    const float* d3w = (const float*)d_in[39];
    const float* t2w = (const float*)d_in[40];
    const float* t2b = (const float*)d_in[41];
    const float* pw  = (const float*)d_in[42];
    const float* pb  = (const float*)d_in[43];

    constexpr long long SZ_X = 589824LL;     // (4,16,96,96)
    constexpr long long SZ_R = 18874368LL;   // (2,32,32,96,96)
    constexpr long long SZ_M = 4718592LL;    // (2,64,16,48,48)
    constexpr long long SZ_S = 1179648LL;    // (2,128,8,24,24)
    constexpr long long SZ_T = 294912LL;     // (2,256,4,12,12)
    float* ws = (float*)d_ws;
    float* X1 = ws;                float* X2 = X1 + SZ_X;
    float* stats = X2 + SZ_X;
    float* R0 = stats + 1024;      float* R1 = R0 + SZ_R;
    float* R2 = R1 + SZ_R;         float* R3 = R2 + SZ_R;
    float* M0 = R3 + SZ_R;         float* M1 = M0 + SZ_M;
    float* Sm0 = M1 + SZ_M;        float* Sm1 = Sm0 + SZ_S;
    float* T0 = Sm1 + SZ_S;        float* T1 = T0 + SZ_T;
    __bf16* pan = (__bf16*)(T1 + SZ_T);   // bf16 weight-panel arena

    auto conv = [&](const float* X, const float* Wf, const float* bias, float* Y,
                    int N, int Cin, int Din, int Hin, int Win, int Cout, int G,
                    int kd, int kh, int kw, int stride,
                    int pd, int ph, int pw_, int dil) {
        int cing = Cin / G, coutg = Cout / G;
        int Kreal = cing * kd * kh * kw;
        int Ksteps = (Kreal + 31) / 32;
        int Ntiles = (coutg + 15) / 16;
        int Do, Ho, Wo;
        if (dil) { Do = 2 * Din; Ho = 2 * Hin; Wo = 2 * Win; }
        else {
            Do = (Din + 2 * pd - kd) / stride + 1;
            Ho = (Hin + 2 * ph - kh) / stride + 1;
            Wo = (Win + 2 * pw_ - kw) / stride + 1;
        }
        int panelElems = G * Ntiles * Ksteps * 512;
        pack_w_kernel<<<idiv(panelElems, 256), 256, 0, stream>>>(
            Wf, pan, G, coutg, cing, kd * kh * kw, kd, kh, kw,
            Kreal, Ksteps, Ntiles, panelElems);
        ConvP p;
        p.N = N; p.Cin = Cin; p.Din = Din; p.Hin = Hin; p.Win = Win;
        p.Cout = Cout; p.Do_ = Do; p.Ho = Ho; p.Wo = Wo;
        p.G = G; p.coutg = coutg; p.cing = cing;
        p.Kreal = Kreal; p.Ksteps = Ksteps; p.Ntiles = Ntiles;
        p.pd = pd; p.ph = ph; p.pw = pw_;
        p.M = (long long)N * Do * Ho * Wo;
        dim3 grid((unsigned)idiv(p.M, 64), (unsigned)G, (unsigned)Ntiles);
        if (dil)
            conv_wmma_kernel<3,3,3,1,1><<<grid, 128, 0, stream>>>(X, pan, bias, Y, p);
        else if (kd == 3 && stride == 2)
            conv_wmma_kernel<3,3,3,2,0><<<grid, 128, 0, stream>>>(X, pan, bias, Y, p);
        else if (kd == 3)
            conv_wmma_kernel<3,3,3,1,0><<<grid, 128, 0, stream>>>(X, pan, bias, Y, p);
        else if (kh == 3)
            conv_wmma_kernel<1,3,3,1,0><<<grid, 128, 0, stream>>>(X, pan, bias, Y, p);
        else
            conv_wmma_kernel<1,1,1,1,0><<<grid, 128, 0, stream>>>(X, pan, bias, Y, p);
        pan += panelElems;
    };

    auto bnrelu = [&](float* X, const float* gamma, const float* beta,
                      int C, int N, long long DHW) {
        bn_stats_kernel<<<C, 256, 0, stream>>>(X, stats, C, N, DHW);
        long long total = (long long)N * C * DHW;
        bn_apply_relu_kernel<<<idiv(total, 256), 256, 0, stream>>>(
            X, stats, gamma, beta, C, DHW, total);
    };

    // ---- reduce ----------------------------------------------------------
    conv(conv_out, r_w1, r_b1, X1, 4, 512, 1, 96, 96, 16, 1, 1, 1, 1, 1, 0, 0, 0, 0);
    conv(X1, r_w2, r_b2, X2, 4, 16, 1, 96, 96, 16, 1, 1, 3, 3, 1, 0, 1, 1, 0);
    bnrelu(X2, r_g, r_bt, 16, 4, 9216LL);

    // ---- cost volume -----------------------------------------------------
    cost_volume_kernel<<<idiv(SZ_R, 256), 256, 0, stream>>>(X2, R0);

    // ---- stacks ----------------------------------------------------------
    conv(R0, S[0][0], nullptr, R1, 2, 32, 32, 96, 96, 32, 8, 3, 3, 3, 1, 1, 1, 1, 0);
    bnrelu(R1, S[0][1], S[0][2], 32, 2, 32LL * 9216);
    conv(R1, S[0][3], nullptr, R2, 2, 32, 32, 96, 96, 32, 8, 3, 3, 3, 1, 1, 1, 1, 0);
    bnrelu(R2, S[0][4], S[0][5], 32, 2, 32LL * 9216);

    conv(R2, S[1][0], nullptr, R0, 2, 32, 32, 96, 96, 32, 8, 3, 3, 3, 1, 1, 1, 1, 0);
    bnrelu(R0, S[1][1], S[1][2], 32, 2, 32LL * 9216);
    conv(R0, S[1][3], nullptr, R3, 2, 32, 32, 96, 96, 32, 8, 3, 3, 3, 1, 1, 1, 1, 0);
    bnrelu(R3, S[1][4], S[1][5], 32, 2, 32LL * 9216);           // s1 kept (skip)

    conv(R3, S[2][0], nullptr, M0, 2, 32, 32, 96, 96, 64, 8, 3, 3, 3, 2, 1, 1, 1, 0);
    bnrelu(M0, S[2][1], S[2][2], 64, 2, 16LL * 48 * 48);
    conv(M0, S[2][3], nullptr, M1, 2, 64, 16, 48, 48, 64, 8, 3, 3, 3, 1, 1, 1, 1, 0);
    bnrelu(M1, S[2][4], S[2][5], 64, 2, 16LL * 48 * 48);        // s1_1 kept

    conv(M1, S[3][0], nullptr, Sm0, 2, 64, 16, 48, 48, 128, 8, 3, 3, 3, 2, 1, 1, 1, 0);
    bnrelu(Sm0, S[3][1], S[3][2], 128, 2, 8LL * 24 * 24);
    conv(Sm0, S[3][3], nullptr, Sm1, 2, 128, 8, 24, 24, 128, 8, 3, 3, 3, 1, 1, 1, 1, 0);
    bnrelu(Sm1, S[3][4], S[3][5], 128, 2, 8LL * 24 * 24);       // s1_2 kept

    conv(Sm1, S[4][0], nullptr, T0, 2, 128, 8, 24, 24, 256, 8, 3, 3, 3, 2, 1, 1, 1, 0);
    bnrelu(T0, S[4][1], S[4][2], 256, 2, 4LL * 12 * 12);
    conv(T0, S[4][3], nullptr, T1, 2, 256, 4, 12, 12, 256, 8, 3, 3, 3, 1, 1, 1, 1, 0);
    bnrelu(T1, S[4][4], S[4][5], 256, 2, 4LL * 12 * 12);        // s1_3

    // ---- decoder (deconv + skip) ----------------------------------------
    conv(T1, d1w, nullptr, Sm0, 2, 256, 4, 12, 12, 128, 8, 3, 3, 3, 1, 1, 1, 1, 1);
    add_kernel<<<idiv(SZ_S, 256), 256, 0, stream>>>(Sm0, Sm1, SZ_S);   // u1
    conv(Sm0, d2w, nullptr, M0, 2, 128, 8, 24, 24, 64, 8, 3, 3, 3, 1, 1, 1, 1, 1);
    add_kernel<<<idiv(SZ_M, 256), 256, 0, stream>>>(M0, M1, SZ_M);     // u2
    conv(M0, d3w, nullptr, R1, 2, 64, 16, 48, 48, 32, 8, 3, 3, 3, 1, 1, 1, 1, 1);
    add_kernel<<<idiv(SZ_R, 256), 256, 0, stream>>>(R1, R3, SZ_R);     // u3

    // ---- heads -----------------------------------------------------------
    float* outp = (float*)d_out;
    to2d_kernel<<<idiv(589824, 256), 256, 0, stream>>>(R1, t2w, t2b, outp);
    predict_kernel<<<idiv(18432, 256), 256, 0, stream>>>(outp, pw, pb, outp + 589824);
}
